// GroupedQueryAttention_5970004541618
// MI455X (gfx1250) — matrix operand order repaired and checked by reading, not compiled, not run on hardware
//
#include <hip/hip_runtime.h>

typedef __bf16 bf16;
typedef __attribute__((ext_vector_type(8)))  __bf16 v8bf;
typedef __attribute__((ext_vector_type(16))) __bf16 v16bf;
typedef __attribute__((ext_vector_type(8)))  float  v8f;
typedef __attribute__((ext_vector_type(4)))  float  v4f;

#define WMMA_BF16(A,B,C) \
  __builtin_amdgcn_wmma_f32_16x16x32_bf16(false,(A),false,(B),(short)0,(C),false,false)

__device__ __forceinline__ v16bf cat16(v8bf lo, v8bf hi) {
  return __builtin_shufflevector(lo, hi, 0,1,2,3,4,5,6,7,8,9,10,11,12,13,14,15);
}

// A-frag 16x32 bf16: lane = 16h + m ; elems e<8 -> K=8h+e ; e>=8 -> K=16+8h+(e-8)
__device__ __forceinline__ v16bf load_fragA(const bf16* row, int h) {
  v8bf lo = *(const v8bf*)(row + 8*h);
  v8bf hi = *(const v8bf*)(row + 16 + 8*h);
  return cat16(lo, hi);
}
// B-frag 32x16 bf16: lane = 16h + n ; elems e<8 -> K=8h+e ; e>=8 -> K=16+8h+(e-8)
__device__ __forceinline__ v16bf load_fragB(const bf16* row, int h) {
  v8bf lo = *(const v8bf*)(row + 8*h);
  v8bf hi = *(const v8bf*)(row + 16 + 8*h);
  return cat16(lo, hi);
}

// Async global -> LDS, 16 bytes per lane, tracked by ASYNCcnt.
__device__ __forceinline__ void async_ld_b128(unsigned lds_byte_off, const void* g) {
  asm volatile("global_load_async_to_lds_b128 %0, %1, off"
               :: "v"(lds_byte_off), "v"((unsigned long long)g)
               : "memory");
}
__device__ __forceinline__ void wait_async0() {
  asm volatile("s_wait_asynccnt 0" ::: "memory");
}
__device__ __forceinline__ unsigned lds_off(const void* p) {
  return (unsigned)(unsigned long long)p;  // low 32 bits = LDS offset (ISA 10.2)
}

// ---------------------------------------------------------------- convert ----
__global__ __launch_bounds__(256) void cvt8_f32_bf16(const float* __restrict__ s,
                                                     bf16* __restrict__ d, long n) {
  long i = ((long)blockIdx.x * 256 + threadIdx.x) * 8;
  if (i >= n) return;
  v4f f0 = *(const v4f*)(s + i);
  v4f f1 = *(const v4f*)(s + i + 4);
  v8bf o;
  #pragma unroll
  for (int e = 0; e < 4; ++e) { o[e] = (bf16)f0[e]; o[4 + e] = (bf16)f1[e]; }
  *(v8bf*)(d + i) = o;
}

// ------------------------------------------------------------------- GEMM ----
// Y[m,n] = sum_k X[m,k] * W[n,k]   (X: MxK row-major, W: NxK row-major)
// Block 128x128, K-step 32, double-buffered async global->LDS staging,
// main loop manually unrolled over the two buffers (64 K per iteration).
// 8 waves = 4(M) x 2(N); wave tile 32x64 = 2x4 fragments.
template<int OUTF32>
__global__ __launch_bounds__(256) void gemm_xwT(const bf16* __restrict__ X,
                                                const bf16* __restrict__ W,
                                                float* __restrict__ Yf,
                                                bf16* __restrict__ Yb,
                                                int M, int N, int K) {
  __shared__ __align__(16) bf16 Xs[2][128][48];   // 32 + 16 pad, 96B rows
  __shared__ __align__(16) bf16 Ws[2][128][48];

  const int tid  = threadIdx.x;
  const int lane = tid & 31;
  const int wid  = tid >> 5;
  const int lrow = lane & 15;
  const int h    = lane >> 4;
  const int wm   = (wid & 3) * 32;
  const int wn   = (wid >> 2) * 64;
  const long mBase = (long)blockIdx.y * 128;
  const long nBase = (long)blockIdx.x * 128;

  v8f acc[2][4];
  #pragma unroll
  for (int i = 0; i < 2; ++i)
    #pragma unroll
    for (int j = 0; j < 4; ++j) acc[i][j] = (v8f){};

  // per-thread staging assignment: row = tid>>1 (0..127), 16-elem half
  const int lrw   = tid >> 1;
  const int lhalf = (tid & 1) * 16;
  const bf16* gxr = X + (mBase + lrw) * K + lhalf;
  const bf16* gwr = W + (nBase + lrw) * K + lhalf;
  const unsigned xo = lds_off(&Xs[0][lrw][lhalf]);
  const unsigned wo = lds_off(&Ws[0][lrw][lhalf]);
  const unsigned bufB = (unsigned)sizeof(Xs[0]);   // 12288 bytes per buffer

  // prefetch first K-tile into buffer 0
  async_ld_b128(xo,      gxr);
  async_ld_b128(xo + 16, gxr + 8);
  async_ld_b128(wo,      gwr);
  async_ld_b128(wo + 16, gwr + 8);

  for (int k0 = 0; k0 < K; k0 += 64) {
    // ---- phase A: consume buffer 0, prefetch k0+32 into buffer 1 ----
    wait_async0();
    __syncthreads();
    {
      async_ld_b128(xo + bufB,      gxr + k0 + 32);
      async_ld_b128(xo + bufB + 16, gxr + k0 + 40);
      async_ld_b128(wo + bufB,      gwr + k0 + 32);
      async_ld_b128(wo + bufB + 16, gwr + k0 + 40);
    }
    {
      v16bf a0 = load_fragA(&Xs[0][wm + lrow][0], h);
      v16bf a1 = load_fragA(&Xs[0][wm + 16 + lrow][0], h);
      #pragma unroll
      for (int j = 0; j < 4; ++j) {
        v16bf bfr = load_fragB(&Ws[0][wn + j * 16 + lrow][0], h);
        acc[0][j] = WMMA_BF16(a0, bfr, acc[0][j]);
        acc[1][j] = WMMA_BF16(a1, bfr, acc[1][j]);
      }
    }
    // ---- phase B: consume buffer 1, prefetch k0+64 into buffer 0 ----
    wait_async0();
    __syncthreads();
    if (k0 + 64 < K) {
      async_ld_b128(xo,      gxr + k0 + 64);
      async_ld_b128(xo + 16, gxr + k0 + 72);
      async_ld_b128(wo,      gwr + k0 + 64);
      async_ld_b128(wo + 16, gwr + k0 + 72);
    }
    {
      v16bf a0 = load_fragA(&Xs[1][wm + lrow][0], h);
      v16bf a1 = load_fragA(&Xs[1][wm + 16 + lrow][0], h);
      #pragma unroll
      for (int j = 0; j < 4; ++j) {
        v16bf bfr = load_fragB(&Ws[1][wn + j * 16 + lrow][0], h);
        acc[0][j] = WMMA_BF16(a0, bfr, acc[0][j]);
        acc[1][j] = WMMA_BF16(a1, bfr, acc[1][j]);
      }
    }
  }

  #pragma unroll
  for (int i = 0; i < 2; ++i)
    #pragma unroll
    for (int j = 0; j < 4; ++j) {
      long col = nBase + wn + j * 16 + lrow;
      #pragma unroll
      for (int r = 0; r < 8; ++r) {
        long m = mBase + wm + i * 16 + r + 8 * h;
        if (OUTF32) Yf[m * N + col] = acc[i][j][r];
        else        Yb[m * N + col] = (bf16)acc[i][j][r];
      }
    }
}

// -------------------------------------------------------------- attention ----
#define T_SEQ 2048
#define CQKV  3072
#define DMODEL 2048

__global__ __launch_bounds__(256) void gqa_attn(const bf16* __restrict__ qkv,
                                                bf16* __restrict__ outb) {
  __shared__ __align__(16) bf16 Ks[32][136];    // keys x dims (272B rows)
  __shared__ __align__(16) bf16 Vt[128][48];    // dims x keys (96B rows)
  __shared__ __align__(16) bf16 Ps[8][16][48];  // per-wave P tile (q x key)

  const int tid  = threadIdx.x;
  const int lane = tid & 31;
  const int wid  = tid >> 5;
  const int lrow = lane & 15;
  const int h    = lane >> 4;

  const int b     = blockIdx.y >> 4;
  const int hq    = blockIdx.y & 15;
  const int kv    = hq >> 2;
  const int qbase = blockIdx.x * 128;

  // Q fragments held in registers for the whole kernel (16 rows x 128 dims)
  const long rowQ = (long)(b * T_SEQ + qbase + wid * 16 + lrow) * CQKV + hq * 128;
  v16bf Qf[4];
  #pragma unroll
  for (int kc = 0; kc < 4; ++kc) Qf[kc] = load_fragA(qkv + rowQ + kc * 32, h);

  v8f O[8];
  #pragma unroll
  for (int d = 0; d < 8; ++d) O[d] = (v8f){};
  float mst[8], lst[8];
  #pragma unroll
  for (int r = 0; r < 8; ++r) { mst[r] = -1e30f; lst[r] = 0.f; }

  const int key = tid >> 3;         // 0..31
  const int seg = (tid & 7) * 16;   // 0..112

  const int kTiles = (qbase + 128) / 32;
  for (int kt = 0; kt < kTiles; ++kt) {
    const int kb = kt * 32;
    __syncthreads();
    {
      const long krow = (long)(b * T_SEQ + kb + key) * CQKV;
      const bf16* gk = qkv + krow + 2048 + kv * 128 + seg;
      *(v8bf*)&Ks[key][seg]     = *(const v8bf*)gk;
      *(v8bf*)&Ks[key][seg + 8] = *(const v8bf*)(gk + 8);
      const bf16* gv = qkv + krow + 2560 + kv * 128 + seg;
      v8bf v0 = *(const v8bf*)gv;
      v8bf v1 = *(const v8bf*)(gv + 8);
      #pragma unroll
      for (int d = 0; d < 8; ++d) Vt[seg + d][key] = v0[d];
      #pragma unroll
      for (int d = 0; d < 8; ++d) Vt[seg + 8 + d][key] = v1[d];
    }
    __syncthreads();

    // S = Q * K^T  (two 16x16 column tiles over the 32-key tile)
    v8f s0 = (v8f){}, s1 = (v8f){};
    #pragma unroll
    for (int kc = 0; kc < 4; ++kc) {
      v16bf bk0 = load_fragB(&Ks[lrow][kc * 32], h);
      v16bf bk1 = load_fragB(&Ks[16 + lrow][kc * 32], h);
      s0 = WMMA_BF16(Qf[kc], bk0, s0);
      s1 = WMMA_BF16(Qf[kc], bk1, s1);
    }

    // scale + causal mask + online softmax (row = r+8h, col = lrow)
    const float sc = 0.08838834764831845f;  // 1/sqrt(128)
    #pragma unroll
    for (int r = 0; r < 8; ++r) {
      int   qr = qbase + wid * 16 + r + 8 * h;
      float v0 = (kb + lrow      <= qr) ? s0[r] * sc : -1e30f;
      float v1 = (kb + 16 + lrow <= qr) ? s1[r] * sc : -1e30f;
      float rmax = fmaxf(v0, v1);
      #pragma unroll
      for (int off = 1; off < 16; off <<= 1)
        rmax = fmaxf(rmax, __shfl_xor(rmax, off, 32));
      float mnew = fmaxf(mst[r], rmax);
      float corr = __expf(mst[r] - mnew);
      float p0 = __expf(v0 - mnew);
      float p1 = __expf(v1 - mnew);
      float rs = p0 + p1;
      #pragma unroll
      for (int off = 1; off < 16; off <<= 1)
        rs += __shfl_xor(rs, off, 32);
      mst[r] = mnew;
      lst[r] = lst[r] * corr + rs;
      #pragma unroll
      for (int d = 0; d < 8; ++d) O[d][r] *= corr;
      Ps[wid][r + 8 * h][lrow]      = (bf16)p0;
      Ps[wid][r + 8 * h][16 + lrow] = (bf16)p1;
    }
    // per-wave C-layout -> A-layout transpose through LDS; wave-local wait
    asm volatile("s_wait_dscnt 0" ::: "memory");

    v16bf pf = load_fragA(&Ps[wid][lrow][0], h);
    #pragma unroll
    for (int d = 0; d < 8; ++d) {
      v16bf vf = load_fragB(&Vt[d * 16 + lrow][0], h);
      O[d] = WMMA_BF16(pf, vf, O[d]);
    }
  }

  #pragma unroll
  for (int d = 0; d < 8; ++d)
    #pragma unroll
    for (int r = 0; r < 8; ++r) {
      int qr = qbase + wid * 16 + r + 8 * h;
      float o = O[d][r] / lst[r];
      outb[(long)(b * T_SEQ + qr) * DMODEL + hq * 128 + d * 16 + lrow] = (bf16)o;
    }
}

// ----------------------------------------------------------------- launch ----
extern "C" void kernel_launch(void* const* d_in, const int* in_sizes, int n_in,
                              void* d_out, int out_size, void* d_ws, size_t ws_size,
                              hipStream_t stream) {
  const float* x  = (const float*)d_in[0];
  const float* wq = (const float*)d_in[1];
  const float* wk = (const float*)d_in[2];
  const float* wv = (const float*)d_in[3];
  const float* wo = (const float*)d_in[4];

  // workspace layout (bf16), ~76 MiB total
  bf16* xb   = (bf16*)d_ws;                        // 4096 x 2048
  bf16* wqkv = xb   + (size_t)4096 * 2048;         // 3072 x 2048 (wq;wk;wv)
  bf16* wob  = wqkv + (size_t)3072 * 2048;         // 2048 x 2048
  bf16* qkv  = wob  + (size_t)2048 * 2048;         // 4096 x 3072
  bf16* attn = qkv  + (size_t)4096 * 3072;         // 4096 x 2048

  auto cv = [&](const float* s, bf16* d, long n) {
    unsigned g = (unsigned)((n / 8 + 255) / 256);
    cvt8_f32_bf16<<<g, 256, 0, stream>>>(s, d, n);
  };
  cv(x,  xb,                           (long)4096 * 2048);
  cv(wq, wqkv,                         (long)2048 * 2048);
  cv(wk, wqkv + (size_t)2048 * 2048,   (long)512  * 2048);
  cv(wv, wqkv + (size_t)2560 * 2048,   (long)512  * 2048);
  cv(wo, wob,                          (long)2048 * 2048);

  // fused QKV projection: (4096 x 2048) x (3072 x 2048)^T -> 4096 x 3072 bf16
  gemm_xwT<0><<<dim3(3072 / 128, 4096 / 128), 256, 0, stream>>>(
      xb, wqkv, nullptr, qkv, 4096, 3072, 2048);

  // flash attention: grid (q-tiles=16, B*H=32)
  gqa_attn<<<dim3(16, 32), 256, 0, stream>>>(qkv, attn);

  // output projection: (4096 x 2048) x (2048 x 2048)^T -> fp32 out
  gemm_xwT<1><<<dim3(2048 / 128, 4096 / 128), 256, 0, stream>>>(
      attn, wob, (float*)d_out, nullptr, 4096, 2048, 2048);
}
